// DeepArcNet_30391188587335
// MI455X (gfx1250) — matrix operand order repaired and checked by reading, not compiled, not run on hardware
//
#include <hip/hip_runtime.h>
#include <hip/hip_bf16.h>

// ---------------------------------------------------------------------------
// Fused DeepArcNet forward for MI455X (gfx1250, wave32, WMMA).
//
// Kernel 1 (prep_weights): pack all weights into d_ws as f16, zero-padded to
//   WMMA-friendly [Npad][Kpad] (k-major) so B fragments are single unguarded
//   32B vector loads.
// Kernel 2 (deeparc_fused): one wave (32 lanes) per workgroup = 8 samples =
//   48 rows of the [rows,68] token matrix staged in LDS with zero-padded
//   strides; every projection runs through v_wmma_f32_16x16x32_f16
//   (A: 3 M-tiles hoisted to registers, each B fragment feeds 3 WMMAs);
//   conv / softmax / layernorm are VALU over LDS.
// ---------------------------------------------------------------------------

typedef _Float16 half_t;
typedef half_t v16h __attribute__((ext_vector_type(16)));
typedef float  v8f  __attribute__((ext_vector_type(8)));
typedef float  v4f  __attribute__((ext_vector_type(4)));

#define SAMPLES_PER_WG 8
#define ROWS       48    // 8 samples * 6 tokens
#define HSTRIDE    100   // padded stride for 68 cols (bank-spread, 16B aligned)
#define QSTRIDE    36    // q/k/v per-head stride (34 cols)
#define FSTRIDE    68    // ff intermediate stride (34 cols + zero pad to 64+)
#define FLATSTRIDE 420   // flattened 408 cols + zero pad
#define EPS        1e-5f

// ---- packed-weight workspace layout (offsets in f16 elements) ----
// QKV : [l][mat(q,k,v)][head][48][96]
#define OFF_QKV  0
#define SZ_QKV   (2 * 3 * 2 * 48 * 96)        // 55296
#define OFF_PROJ (OFF_QKV + SZ_QKV)           // [l][80][96]
#define SZ_PROJ  (2 * 80 * 96)                // 15360
#define OFF_FF1  (OFF_PROJ + SZ_PROJ)         // [l][48][96]
#define SZ_FF1   (2 * 48 * 96)                // 9216
#define OFF_FF2  (OFF_FF1 + SZ_FF1)           // [l][80][64]
#define SZ_FF2   (2 * 80 * 64)                // 10240
#define OFF_FC   (OFF_FF2 + SZ_FF2)           // [16][416]
#define SZ_FC    (16 * 416)                   // 6656
#define WP_TOTAL (OFF_FC + SZ_FC)             // 96768 f16 = 193536 bytes

// ===========================================================================
// Weight packing kernel: B[k][n] = W[n*sK.. ] -> wp[region][n][k] f16, padded.
// ===========================================================================
__global__ __launch_bounds__(256) void prep_weights(
    const float* __restrict__ wq, const float* __restrict__ wk,
    const float* __restrict__ wv, const float* __restrict__ proj_w,
    const float* __restrict__ ff1_w, const float* __restrict__ ff2_w,
    const float* __restrict__ fc_w, half_t* __restrict__ wp) {
  const int i = blockIdx.x * 256 + threadIdx.x;
  if (i >= WP_TOTAL) return;
  float v = 0.0f;
  if (i < OFF_PROJ) {                       // QKV
    int j = i;
    const int k = j % 96; j /= 96;
    const int n = j % 48; j /= 48;
    const int head = j % 2; j /= 2;
    const int mat = j % 3;
    const int l = j / 3;
    if (n < 34 && k < 68) {
      const float* W = (mat == 0) ? wq : ((mat == 1) ? wk : wv);
      v = W[(((size_t)l * 2 + head) * 34 + n) * 68 + k];
    }
  } else if (i < OFF_FF1) {                 // PROJ
    int j = i - OFF_PROJ;
    const int k = j % 96; j /= 96;
    const int n = j % 80;
    const int l = j / 80;
    if (n < 68 && k < 68) v = proj_w[((size_t)l * 68 + n) * 68 + k];
  } else if (i < OFF_FF2) {                 // FF1
    int j = i - OFF_FF1;
    const int k = j % 96; j /= 96;
    const int n = j % 48;
    const int l = j / 48;
    if (n < 34 && k < 68) v = ff1_w[((size_t)l * 34 + n) * 68 + k];
  } else if (i < OFF_FC) {                  // FF2 (B[k][n] = ff2_w[l][n][k])
    int j = i - OFF_FF2;
    const int k = j % 64; j /= 64;
    const int n = j % 80;
    const int l = j / 80;
    if (n < 68 && k < 34) v = ff2_w[((size_t)l * 68 + n) * 34 + k];
  } else {                                  // FC
    int j = i - OFF_FC;
    const int k = j % 416;
    const int n = j / 416;
    if (n < 6 && k < 408) v = fc_w[(size_t)n * 408 + k];
  }
  wp[i] = (half_t)v;
}

// ===========================================================================
// Fragment builders (no per-element guards).
// ===========================================================================
__device__ __forceinline__ v8f wmma_f16(v16h a, v16h b, v8f c) {
  return __builtin_amdgcn_wmma_f32_16x16x32_f16(false, a, false, b,
                                                (short)0, c, false, false);
}

// A 16x32 f16 fragment from row-major f32 LDS (zero-padded past logical K).
// lane 0-15 : M=lane,    halves 0-15 -> K=kb+0..15
// lane 16-31: M=lane-16, halves 0-15 -> K=kb+16..31
__device__ __forceinline__ v16h load_a_frag(const float* Abuf, int lda, int r0,
                                            int kbase, int lane) {
  const int m  = lane & 15;
  const int kh = (lane >> 4) * 16;
  const float* p = Abuf + (r0 + m) * lda + kbase + kh;
  const v4f x0 = *(const v4f*)(p);
  const v4f x1 = *(const v4f*)(p + 4);
  const v4f x2 = *(const v4f*)(p + 8);
  const v4f x3 = *(const v4f*)(p + 12);
  v16h a;
#pragma unroll
  for (int i = 0; i < 4; ++i) {
    a[i]      = (half_t)x0[i];
    a[4 + i]  = (half_t)x1[i];
    a[8 + i]  = (half_t)x2[i];
    a[12 + i] = (half_t)x3[i];
  }
  return a;
}

// B 32x16 f16 fragment from packed global weights: one 32B contiguous load.
__device__ __forceinline__ v16h load_b_frag(const half_t* __restrict__ Wp,
                                            int Kpad, int n0, int kbase,
                                            int lane) {
  const half_t* p =
      Wp + (size_t)(n0 + (lane & 15)) * Kpad + kbase + (lane >> 4) * 16;
  return *(const v16h*)p;
}

// GEMM over all 48 rows: dst[row,n] = epi( sum_k Abuf[row,k] * Bpad[k][n] ).
// All 3x KSTEPS A fragments hoisted; each B fragment feeds 3 WMMAs.
// EPI: 0 plain; 1 acc + bias[n] + res[row*HSTRIDE+n]; 2 relu(acc + bias[n])
template <int KSTEPS, int NTILES, int NDIM, int EPI>
__device__ __forceinline__ void gemm_rows48(const float* Abuf, int lda,
                                            const half_t* __restrict__ Wp,
                                            int Kpad, float* dst, int ldd,
                                            const float* bias,
                                            const float* res, int lane) {
  v16h afr[3][KSTEPS];
#pragma unroll
  for (int mt = 0; mt < 3; ++mt)
#pragma unroll
    for (int ks = 0; ks < KSTEPS; ++ks)
      afr[mt][ks] = load_a_frag(Abuf, lda, mt * 16, ks * 32, lane);
#pragma unroll
  for (int nt = 0; nt < NTILES; ++nt) {
    const int n0 = nt * 16;
    v8f acc[3];
#pragma unroll
    for (int mt = 0; mt < 3; ++mt)
      acc[mt] = (v8f){0.f, 0.f, 0.f, 0.f, 0.f, 0.f, 0.f, 0.f};
#pragma unroll
    for (int ks = 0; ks < KSTEPS; ++ks) {
      const v16h b = load_b_frag(Wp, Kpad, n0, ks * 32, lane);
      acc[0] = wmma_f16(afr[0][ks], b, acc[0]);
      acc[1] = wmma_f16(afr[1][ks], b, acc[1]);
      acc[2] = wmma_f16(afr[2][ks], b, acc[2]);
    }
    const int n = n0 + (lane & 15);
    if (n < NDIM) {
      const int mb = (lane >> 4) * 8;
#pragma unroll
      for (int mt = 0; mt < 3; ++mt)
#pragma unroll
        for (int r = 0; r < 8; ++r) {
          const int row = mt * 16 + mb + r;
          float v = acc[mt][r];
          if (EPI == 1) v = v + bias[n] + res[row * HSTRIDE + n];
          if (EPI == 2) v = fmaxf(v + bias[n], 0.0f);
          dst[row * ldd + n] = v;
        }
    }
  }
}

// ===========================================================================
// VALU pieces.
// ===========================================================================
__device__ __forceinline__ void zero_region(float* p, int nvec4, int lane) {
  const v4f z = {0.f, 0.f, 0.f, 0.f};
  for (int i = lane; i < nvec4; i += 32) ((v4f*)p)[i] = z;
}

// conv(5x5,s2,valid,depthwise) + relu + Linear(1,4) + sinusoid -> hb[48x68]
__device__ __forceinline__ void conv_embed(const float* __restrict__ x,
                                           const float* __restrict__ cw,
                                           const float* __restrict__ cb,
                                           const float* __restrict__ lw,
                                           const float* __restrict__ lb,
                                           float* hb, int b0, int Bsz,
                                           int lane) {
  const float fr1 = 1.0f / 18.0f;  // exp(-log(18))
  for (int idx = lane; idx < SAMPLES_PER_WG * 6 * 17; idx += 32) {
    const int s   = idx / 102;
    const int rem = idx % 102;
    const int c   = rem / 17;
    const int t   = rem % 17;
    float acc = cb[c];
    if (b0 + s < Bsz) {
      const float* xp = x + (((size_t)(b0 + s) * 6 + c) * 37 + 2 * t) * 5;
      const float* wp = cw + c * 25;
#pragma unroll
      for (int i = 0; i < 5; ++i)
#pragma unroll
        for (int j = 0; j < 5; ++j) acc += xp[i * 5 + j] * wp[i * 5 + j];
    }
    acc = fmaxf(acc, 0.0f);
    const float tf = (float)t;
    float pos[4];
    pos[0] = __sinf(tf);
    pos[1] = __sinf(tf * fr1);
    pos[2] = __cosf(tf);
    pos[3] = __cosf(tf * fr1);
    float* hrow = hb + (s * 6 + c) * HSTRIDE;
#pragma unroll
    for (int k2 = 0; k2 < 4; ++k2)
      hrow[t * 4 + k2] = acc * lw[c * 4 + k2] + lb[c * 4 + k2] + pos[k2];
  }
}

// per-head attention: softmax(q k^T / sqrt(34)) v -> yb cols [col0, col0+34)
__device__ __forceinline__ void attention48(const float* qb, const float* kb,
                                            const float* vb, float* yb,
                                            int col0, int lane) {
  const float iscale = rsqrtf(34.0f);
  for (int idx = lane; idx < ROWS; idx += 32) {
    const int s = idx / 6;
    const int t = idx % 6;
    const float* qr = qb + (s * 6 + t) * QSTRIDE;
    float sc[6];
    float mx = -1e30f;
#pragma unroll
    for (int s2 = 0; s2 < 6; ++s2) {
      const float* kr = kb + (s * 6 + s2) * QSTRIDE;
      float d = 0.0f;
      for (int e = 0; e < 34; ++e) d += qr[e] * kr[e];
      d *= iscale;
      sc[s2] = d;
      mx = fmaxf(mx, d);
    }
    float sum = 0.0f;
#pragma unroll
    for (int s2 = 0; s2 < 6; ++s2) {
      sc[s2] = __expf(sc[s2] - mx);
      sum += sc[s2];
    }
    const float inv = 1.0f / sum;
    float* yrow = yb + (s * 6 + t) * HSTRIDE + col0;
    for (int e = 0; e < 34; ++e) {
      float o = 0.0f;
#pragma unroll
      for (int s2 = 0; s2 < 6; ++s2)
        o += sc[s2] * vb[(s * 6 + s2) * QSTRIDE + e];
      yrow[e] = o * inv;
    }
  }
}

__device__ __forceinline__ void layernorm48(const float* src, float* dst,
                                            const float* g, const float* bt,
                                            int lane) {
  for (int row = lane; row < ROWS; row += 32) {
    const float* r = src + row * HSTRIDE;
    float m = 0.0f;
    for (int e = 0; e < 68; ++e) m += r[e];
    m *= (1.0f / 68.0f);
    float v = 0.0f;
    for (int e = 0; e < 68; ++e) {
      const float d = r[e] - m;
      v += d * d;
    }
    v *= (1.0f / 68.0f);
    const float inv = rsqrtf(v + EPS);
    float* o = dst + row * HSTRIDE;
    for (int e = 0; e < 68; ++e) o[e] = (r[e] - m) * inv * g[e] + bt[e];
  }
}

// final layernorm, written to flattened zero-padded [8][FLATSTRIDE] buffer
__device__ __forceinline__ void layernorm_flat(const float* src, float* flat,
                                               const float* g, const float* bt,
                                               int lane) {
  for (int row = lane; row < ROWS; row += 32) {
    const float* r = src + row * HSTRIDE;
    float m = 0.0f;
    for (int e = 0; e < 68; ++e) m += r[e];
    m *= (1.0f / 68.0f);
    float v = 0.0f;
    for (int e = 0; e < 68; ++e) {
      const float d = r[e] - m;
      v += d * d;
    }
    v *= (1.0f / 68.0f);
    const float inv = rsqrtf(v + EPS);
    const int s = row / 6;
    const int c = row % 6;
    float* o = flat + s * FLATSTRIDE + c * 68;
    for (int e = 0; e < 68; ++e) o[e] = (r[e] - m) * inv * g[e] + bt[e];
  }
}

// fc: out[b,n] = relu(sum_{k<408} flat[b][k] * fcB[k][n] + fc_b[n]); N=6, M=8
__device__ __forceinline__ void fc_out(const float* flat,
                                       const half_t* __restrict__ Wfc,
                                       const float* __restrict__ fcb,
                                       float* __restrict__ out, int b0, int Bsz,
                                       int lane) {
  const int m    = lane & 15;
  const int bsel = lane >> 4;
  const int kh   = bsel * 16;
  v8f acc = {0.f, 0.f, 0.f, 0.f, 0.f, 0.f, 0.f, 0.f};
#pragma unroll
  for (int ks = 0; ks < 13; ++ks) {
    v16h a;
    if (m < SAMPLES_PER_WG) {
      const float* p = flat + m * FLATSTRIDE + ks * 32 + kh;
      const v4f x0 = *(const v4f*)(p);
      const v4f x1 = *(const v4f*)(p + 4);
      const v4f x2 = *(const v4f*)(p + 8);
      const v4f x3 = *(const v4f*)(p + 12);
#pragma unroll
      for (int i = 0; i < 4; ++i) {
        a[i]      = (half_t)x0[i];
        a[4 + i]  = (half_t)x1[i];
        a[8 + i]  = (half_t)x2[i];
        a[12 + i] = (half_t)x3[i];
      }
    } else {
#pragma unroll
      for (int i = 0; i < 16; ++i) a[i] = (half_t)0.0f;
    }
    const v16h b = load_b_frag(Wfc, 416, 0, ks * 32, lane);
    acc = wmma_f16(a, b, acc);
  }
  const int n = lane & 15;
  if (n < 6) {
#pragma unroll
    for (int r = 0; r < 8; ++r) {
      const int mm = r + 8 * bsel;  // C row = sample index
      if (mm < SAMPLES_PER_WG && (b0 + mm) < Bsz)
        out[(size_t)(b0 + mm) * 6 + n] = fmaxf(acc[r] + fcb[n], 0.0f);
    }
  }
}

// ===========================================================================
// Fused kernel.
// ===========================================================================
__global__ __launch_bounds__(32) void deeparc_fused(
    const float* __restrict__ x, const float* __restrict__ conv_w,
    const float* __restrict__ conv_b, const float* __restrict__ lemb_w,
    const float* __restrict__ lemb_b, const half_t* __restrict__ wp,
    const float* __restrict__ proj_b, const float* __restrict__ ff1_b,
    const float* __restrict__ ff2_b, const float* __restrict__ ln1_g,
    const float* __restrict__ ln1_b, const float* __restrict__ ln2_g,
    const float* __restrict__ ln2_b, const float* __restrict__ lnf_g,
    const float* __restrict__ lnf_b, const float* __restrict__ fc_b,
    float* __restrict__ out, int Bsz) {
  __shared__ __align__(32) float smem[2 * ROWS * HSTRIDE + 3 * ROWS * QSTRIDE];
  float* hb = smem;                       // 48 x 100
  float* yb = smem + ROWS * HSTRIDE;      // 48 x 100
  float* qb = yb + ROWS * HSTRIDE;        // 48 x 36
  float* kb = qb + ROWS * QSTRIDE;        // 48 x 36
  float* vb = kb + ROWS * QSTRIDE;        // 48 x 36
  float* fbuf = qb;                       // 48 x 68  (aliases q/k, ff phase)
  float* flat = qb;                       // 8 x 420  (aliases q/k/v, fc phase)

  const half_t* WQKV  = wp + OFF_QKV;
  const half_t* WPROJ = wp + OFF_PROJ;
  const half_t* WFF1  = wp + OFF_FF1;
  const half_t* WFF2  = wp + OFF_FF2;
  const half_t* WFC   = wp + OFF_FC;

  const int lane = threadIdx.x;
  const int b0 = blockIdx.x * SAMPLES_PER_WG;

  // zero hb + yb (establishes the zero-pad invariant in cols 68..99)
  zero_region(smem, 2 * ROWS * HSTRIDE / 4, lane);
  __syncthreads();

  conv_embed(x, conv_w, conv_b, lemb_w, lemb_b, hb, b0, Bsz, lane);
  __syncthreads();

  for (int l = 0; l < 2; ++l) {
    for (int hh = 0; hh < 2; ++hh) {
      const half_t* wq_h = WQKV + ((size_t)(l * 3 + 0) * 2 + hh) * 48 * 96;
      const half_t* wk_h = WQKV + ((size_t)(l * 3 + 1) * 2 + hh) * 48 * 96;
      const half_t* wv_h = WQKV + ((size_t)(l * 3 + 2) * 2 + hh) * 48 * 96;
      gemm_rows48<3, 3, 34, 0>(hb, HSTRIDE, wq_h, 96, qb, QSTRIDE, nullptr,
                               nullptr, lane);
      gemm_rows48<3, 3, 34, 0>(hb, HSTRIDE, wk_h, 96, kb, QSTRIDE, nullptr,
                               nullptr, lane);
      gemm_rows48<3, 3, 34, 0>(hb, HSTRIDE, wv_h, 96, vb, QSTRIDE, nullptr,
                               nullptr, lane);
      __syncthreads();
      attention48(qb, kb, vb, yb, hh * 34, lane);
      __syncthreads();
    }
    // proj + bias + residual -> yb (in place; all A frags register-resident)
    gemm_rows48<3, 5, 68, 1>(yb, HSTRIDE, WPROJ + (size_t)l * 80 * 96, 96, yb,
                             HSTRIDE, proj_b + l * 68, hb, lane);
    __syncthreads();
    layernorm48(yb, hb, ln1_g + l * 68, ln1_b + l * 68, lane);
    zero_region(fbuf, ROWS * FSTRIDE / 4, lane);  // zero-pad ff buffer
    __syncthreads();
    gemm_rows48<3, 3, 34, 2>(hb, HSTRIDE, WFF1 + (size_t)l * 48 * 96, 96, fbuf,
                             FSTRIDE, ff1_b + l * 34, nullptr, lane);
    __syncthreads();
    gemm_rows48<2, 5, 68, 1>(fbuf, FSTRIDE, WFF2 + (size_t)l * 80 * 64, 64, yb,
                             HSTRIDE, ff2_b + l * 68, hb, lane);
    __syncthreads();
    layernorm48(yb, hb, ln2_g + l * 68, ln2_b + l * 68, lane);
    __syncthreads();
  }

  zero_region(flat, SAMPLES_PER_WG * FLATSTRIDE / 4, lane);
  __syncthreads();
  layernorm_flat(hb, flat, lnf_g, lnf_b, lane);
  __syncthreads();
  fc_out(flat, WFC, fc_b, out, b0, Bsz, lane);
}

// ===========================================================================
extern "C" void kernel_launch(void* const* d_in, const int* in_sizes, int n_in,
                              void* d_out, int out_size, void* d_ws,
                              size_t ws_size, hipStream_t stream) {
  (void)n_in; (void)out_size; (void)ws_size;
  const float* x      = (const float*)d_in[0];
  const float* conv_w = (const float*)d_in[1];
  const float* conv_b = (const float*)d_in[2];
  const float* lemb_w = (const float*)d_in[3];
  const float* lemb_b = (const float*)d_in[4];
  const float* wq     = (const float*)d_in[5];
  const float* wk     = (const float*)d_in[6];
  const float* wv     = (const float*)d_in[7];
  const float* proj_w = (const float*)d_in[8];
  const float* proj_b = (const float*)d_in[9];
  const float* ff1_w  = (const float*)d_in[10];
  const float* ff1_b  = (const float*)d_in[11];
  const float* ff2_w  = (const float*)d_in[12];
  const float* ff2_b  = (const float*)d_in[13];
  const float* ln1_g  = (const float*)d_in[14];
  const float* ln1_b  = (const float*)d_in[15];
  const float* ln2_g  = (const float*)d_in[16];
  const float* ln2_b  = (const float*)d_in[17];
  const float* lnf_g  = (const float*)d_in[18];
  const float* lnf_b  = (const float*)d_in[19];
  const float* fc_w   = (const float*)d_in[20];
  const float* fc_b   = (const float*)d_in[21];
  float* out = (float*)d_out;
  half_t* wp = (half_t*)d_ws;  // needs WP_TOTAL*2 = 193536 bytes

  const int Bsz = in_sizes[0] / (6 * 37 * 5);

  prep_weights<<<(WP_TOTAL + 255) / 256, 256, 0, stream>>>(
      wq, wk, wv, proj_w, ff1_w, ff2_w, fc_w, wp);

  const int wgs = (Bsz + SAMPLES_PER_WG - 1) / SAMPLES_PER_WG;
  deeparc_fused<<<wgs, 32, 0, stream>>>(
      x, conv_w, conv_b, lemb_w, lemb_b, wp, proj_b, ff1_b, ff2_b, ln1_g,
      ln1_b, ln2_g, ln2_b, lnf_g, lnf_b, fc_b, out, Bsz);
}
